// LinearAttention3d_22247930593920
// MI455X (gfx1250) — matrix operand order repaired and checked by reading, not compiled, not run on hardware
//
#include <hip/hip_runtime.h>

typedef __attribute__((ext_vector_type(16))) __bf16    v16bf;
typedef __attribute__((ext_vector_type(2)))  __bf16    bf16x2;
typedef __attribute__((ext_vector_type(8)))  float     v8f;
typedef __attribute__((ext_vector_type(2)))  float     f32x2;
typedef __attribute__((ext_vector_type(4)))  unsigned  u32x4;
typedef __attribute__((ext_vector_type(8)))  int       i32x8;
typedef __attribute__((ext_vector_type(4)))  int       i32x4;

constexpr int NTOK  = 32 * 32 * 32;   // 32768 tokens
constexpr int CIN   = 256;            // channels
constexpr int NB    = 2;              // batch
constexpr int MQKV  = 768;            // 3 * inner
constexpr int NH    = 8;
constexpr int DHEAD = 32;

// workspace layout (units: floats)
// q,k logits stay f32 (softmax precision); v is stored bf16 (it is only ever
// consumed as a bf16 WMMA operand, so rounding at the GEMM epilogue is free).
constexpr size_t SZ_QK    = (size_t)NB * 2 * CIN * NTOK;       // 134 MB f32
constexpr size_t OFF_QK   = 0;
constexpr size_t OFF_V    = OFF_QK + SZ_QK;                    // NB*256*NTOK bf16
constexpr size_t SZ_V_F   = (size_t)NB * CIN * NTOK / 2;       // in float units
constexpr size_t OFF_KMAX = OFF_V + SZ_V_F;
constexpr size_t OFF_KSUM = OFF_KMAX + (size_t)NB * CIN;
constexpr size_t OFF_CTX  = OFF_KSUM + (size_t)NB * CIN;
constexpr size_t SZ_CTX   = (size_t)NB * NH * DHEAD * DHEAD;
constexpr size_t OFF_M2   = OFF_CTX + SZ_CTX;

// ---- bf16 WMMA fragment builders (contiguous k = 16g + s in both A and B, wave32) ----

// A-matrix 16x32 bf16: lane (c=l&15,g=l>>4) holds row M=c,
// VGPR j -> K = 2j + 16g (pairs K,K+1 packed lo/hi)
__device__ __forceinline__ v16bf make_a_frag(const float* rowp, int kbase, int g) {
  v16bf a;
#pragma unroll
  for (int j = 0; j < 8; ++j) {
    int   kk = kbase + 2 * j + 16 * g;
    f32x2 v  = *reinterpret_cast<const f32x2*>(rowp + kk);
    a[2 * j]     = (__bf16)v.x;
    a[2 * j + 1] = (__bf16)v.y;
  }
  return a;
}

// B-matrix 32x16 bf16 from a row-major [k][64 n] LDS panel:
// lane holds column N=c; VGPR j -> K = 2j + 16g.
// Lanes 0..15 read consecutive n at fixed k -> conflict-free ds_read_b32.
__device__ __forceinline__ v16bf make_b_frag_rm(const float* panel, int n, int kbase,
                                                int g) {
  v16bf bf;
#pragma unroll
  for (int j = 0; j < 8; ++j) {
    int kk = kbase + 2 * j + 16 * g;
    bf[2 * j]     = (__bf16)panel[kk * 64 + n];
    bf[2 * j + 1] = (__bf16)panel[(kk + 1) * 64 + n];
  }
  return bf;
}

// ---- TDM: one descriptor pulls a [128 k x 64 n] f32 tile into LDS ----
// D# per ISA ch.8: group0 = {count=1 | type=2, lds_addr, global_addr 57b},
// group1 = {data_size=4B, tensor_dim0=NTOK, tensor_dim1=128, tile=64x128,
//           tensor_dim0_stride=NTOK}. 2D tile -> optional groups zero.
__device__ __forceinline__ void tdm_load_panel(const float* gsrc, unsigned lds_byte_off) {
  unsigned long long ga = (unsigned long long)(size_t)gsrc;
  u32x4 g0;
  g0[0] = 1u;                                               // count=1 (user D#)
  g0[1] = lds_byte_off;                                     // lds_addr
  g0[2] = (unsigned)(ga & 0xffffffffull);                   // global_addr[31:0]
  g0[3] = (unsigned)((ga >> 32) & 0x01ffffffull) | (2u << 30);  // [56:32] | type=2
  i32x8 g1;
  g1[0] = (int)(2u << 16);                                  // data_size=2 (4 bytes)
  g1[1] = (int)(((unsigned)NTOK & 0xffffu) << 16);          // tensor_dim0[15:0]
  g1[2] = (int)(((unsigned)NTOK >> 16) | (128u << 16));     // dim0 hi | tensor_dim1=128
  g1[3] = (int)(64u << 16);                                 // tile_dim0=64
  g1[4] = 128;                                              // tile_dim1=128
  g1[5] = NTOK;                                             // tensor_dim0_stride lo32
  g1[6] = 0;
  g1[7] = 0;
  i32x4 gz4 = {0, 0, 0, 0};
  i32x8 gz8 = {0, 0, 0, 0, 0, 0, 0, 0};
  __builtin_amdgcn_tensor_load_to_lds(g0, g1, gz4, gz4, gz8, 0);
}

// ---- Kernel 1 & 5: C[b] = A[b](Mx256) x B[b](256xNTOK) ----
// 64x64 tile / 8-wave block. B panels double-buffered in LDS via TDM; wave 0
// issues both loads up front and gates phases with s_wait_tensorcnt (TDM
// completes in order). QSOFTMAX applies the per-token head softmax to the
// staged panel. Rows >= vsplit (the v part of qkv) are stored as bf16.
template <bool QSOFTMAX>
__global__ void __launch_bounds__(256)
gemm_panel_kernel(const float* __restrict__ A, size_t bstrideA,
                  const float* __restrict__ Bsrc, size_t bstrideB,
                  float* __restrict__ OutF, size_t bstrideOF,
                  __bf16* __restrict__ OutV, size_t bstrideOV, int vsplit) {
  // Single static LDS allocation -> starts at LDS offset 0 (byte offsets below).
  __shared__ float lds[2][128 * 64];

  const int b    = blockIdx.z;
  const int n0   = blockIdx.y * 64;
  const int m0   = blockIdx.x * 64;
  const int t    = threadIdx.x;
  const int lane = t & 31;
  const int wave = t >> 5;
  const int c    = lane & 15;
  const int g    = lane >> 4;
  const int mw   = m0 + (wave >> 1) * 16;  // wave's 16-row M tile
  const int nw   = (wave & 1) * 32;        // wave's 32-col N range

  const float* Bb   = Bsrc + (size_t)b * bstrideB + n0;
  const float* Arow = A + (size_t)b * bstrideA + (size_t)(mw + c) * CIN;

  if (wave == 0) {  // one wave owns the TDM queue (TENSORcnt is per-wave)
    tdm_load_panel(Bb, 0u);
    tdm_load_panel(Bb + (size_t)128 * NTOK, 128u * 64u * 4u);
  }

  v8f acc0 = {};
  v8f acc1 = {};

  // per-phase body (softmax on staged panel + 4 x K=32 WMMA steps)
  auto phase = [&](int ph) {
    const float* panel = &lds[ph][0];
    const int    krow0 = ph * 128;

    if (QSOFTMAX) {
      // softmax over the 32 head-dims of each (token, head); 4 heads per panel
      int   nl = t & 63;
      int   hh = t >> 6;  // 0..3
      float mx = lds[ph][(hh * DHEAD) * 64 + nl];
#pragma unroll
      for (int d = 1; d < DHEAD; ++d)
        mx = fmaxf(mx, lds[ph][(hh * DHEAD + d) * 64 + nl]);
      float s = 0.0f;
#pragma unroll
      for (int d = 0; d < DHEAD; ++d) {
        float e = __expf(lds[ph][(hh * DHEAD + d) * 64 + nl] - mx);
        lds[ph][(hh * DHEAD + d) * 64 + nl] = e;
        s += e;
      }
      float inv = 1.0f / s;
#pragma unroll
      for (int d = 0; d < DHEAD; ++d) lds[ph][(hh * DHEAD + d) * 64 + nl] *= inv;
      __syncthreads();
    }

#pragma unroll
    for (int kb = 0; kb < 4; ++kb) {
      v16bf af  = make_a_frag(Arow, krow0 + kb * 32, g);
      v16bf bf0 = make_b_frag_rm(panel, nw + c, kb * 32, g);
      v16bf bf1 = make_b_frag_rm(panel, nw + 16 + c, kb * 32, g);
      acc0 = __builtin_amdgcn_wmma_f32_16x16x32_bf16(false, af, false, bf0, (short)0,
                                                     acc0, false, false);
      acc1 = __builtin_amdgcn_wmma_f32_16x16x32_bf16(false, af, false, bf1, (short)0,
                                                     acc1, false, false);
    }
  };

  // phase 0: wait until only the phase-1 TDM load is outstanding
  if (wave == 0) __builtin_amdgcn_s_wait_tensorcnt(1);
  __syncthreads();
  phase(0);
  // phase 1: wait for all TDM loads
  if (wave == 0) __builtin_amdgcn_s_wait_tensorcnt(0);
  __syncthreads();
  phase(1);

  // C/D layout: VGPR r -> row M = r + 8g, col N = c (coalesced stores)
  if (m0 < vsplit) {
    float* Ob = OutF + (size_t)b * bstrideOF;
#pragma unroll
    for (int r = 0; r < 8; ++r) {
      int row = mw + r + 8 * g;
      Ob[(size_t)row * NTOK + n0 + nw + c]      = acc0[r];
      Ob[(size_t)row * NTOK + n0 + nw + 16 + c] = acc1[r];
    }
  } else {  // v rows -> bf16 (consumed only as bf16 WMMA operand later)
    __bf16* Ov = OutV + (size_t)b * bstrideOV;
#pragma unroll
    for (int r = 0; r < 8; ++r) {
      int row = mw + r + 8 * g - vsplit;
      Ov[(size_t)row * NTOK + n0 + nw + c]      = (__bf16)acc0[r];
      Ov[(size_t)row * NTOK + n0 + nw + 16 + c] = (__bf16)acc1[r];
    }
  }
}

// ---- Kernel 2: per k-channel max & sum(exp) over all 32768 tokens ----
__global__ void __launch_bounds__(256)
k_reduce_kernel(const float* __restrict__ qk, float* __restrict__ kmax,
                float* __restrict__ ksum) {
  __shared__ float red[256];
  const int    r   = blockIdx.x;  // 0..511 -> (b, channel)
  const int    b   = r >> 8;
  const int    i   = r & 255;
  const float* row = qk + ((size_t)b * 2 * CIN + CIN + i) * NTOK;
  const int    t   = threadIdx.x;

  float mx = -3.4e38f;
  for (int n = t; n < NTOK; n += 256) mx = fmaxf(mx, row[n]);
  red[t] = mx;
  __syncthreads();
  for (int s = 128; s > 0; s >>= 1) {
    if (t < s) red[t] = fmaxf(red[t], red[t + s]);
    __syncthreads();
  }
  mx = red[0];
  __syncthreads();

  float sm = 0.0f;
  for (int n = t; n < NTOK; n += 256) sm += __expf(row[n] - mx);
  red[t] = sm;
  __syncthreads();
  for (int s = 128; s > 0; s >>= 1) {
    if (t < s) red[t] += red[t + s];
    __syncthreads();
  }
  if (t == 0) {
    kmax[r] = mx;
    ksum[r] = red[0];
  }
}

// ---- Kernel 3: context[b,h] (32x32) = softmax_n(k)^T @ v, split-K over n ----
__global__ void __launch_bounds__(128)
context_kernel(const float* __restrict__ qk, const __bf16* __restrict__ vbf,
               const float* __restrict__ kmax, const float* __restrict__ ksum,
               float* __restrict__ ctx, int chunk) {
  const int bh    = blockIdx.y;  // 0..15
  const int b     = bh >> 3;
  const int h     = bh & 7;
  const int nbase = blockIdx.x * chunk;
  const int t     = threadIdx.x;
  const int lane  = t & 31;
  const int wave  = t >> 5;
  const int c     = lane & 15;
  const int g     = lane >> 4;
  const int d0    = (wave >> 1) * 16;
  const int e0    = (wave & 1) * 16;

  const float*  krow = qk + ((size_t)b * 2 * CIN + CIN + h * DHEAD + d0 + c) * NTOK;
  const __bf16* vrow = vbf + ((size_t)b * CIN + h * DHEAD + e0 + c) * NTOK;
  const float   mx   = kmax[b * CIN + h * DHEAD + d0 + c];
  const float   inv  = 1.0f / ksum[b * CIN + h * DHEAD + d0 + c];

  v8f acc = {};
  for (int n = nbase; n < nbase + chunk; n += 32) {
    v16bf af, bfg;
#pragma unroll
    for (int j = 0; j < 8; ++j) {
      int   kk = n + 2 * j + 16 * g;  // A pattern
      f32x2 v  = *reinterpret_cast<const f32x2*>(krow + kk);
      af[2 * j]     = (__bf16)(__expf(v.x - mx) * inv);
      af[2 * j + 1] = (__bf16)(__expf(v.y - mx) * inv);
      int    k2 = n + 2 * j + 16 * g;                   // B pattern (bf16 pair)
      bf16x2 w  = *reinterpret_cast<const bf16x2*>(vrow + k2);
      bfg[2 * j]     = w[0];
      bfg[2 * j + 1] = w[1];
    }
    acc = __builtin_amdgcn_wmma_f32_16x16x32_bf16(false, af, false, bfg, (short)0,
                                                  acc, false, false);
  }
#pragma unroll
  for (int r = 0; r < 8; ++r) {
    int d = d0 + r + 8 * g;
    atomicAdd(&ctx[(((size_t)b * NH + h) * DHEAD + d) * DHEAD + e0 + c], acc[r]);
  }
}

// ---- Kernel 4: M2[b,o,(h,d)] = sum_e Wout[o,(h,e)] * ctx[b,h,d,e] ----
__global__ void __launch_bounds__(256)
fold_kernel(const float* __restrict__ w_out, const float* __restrict__ ctx,
            float* __restrict__ m2) {
  const int idx = blockIdx.x * 256 + threadIdx.x;  // 0..131071
  const int b   = idx >> 16;
  const int o   = (idx >> 8) & 255;
  const int i   = idx & 255;
  const int h   = i >> 5;
  const int d   = i & 31;
  const float* wrow = w_out + (size_t)o * CIN + h * DHEAD;
  const float* cr   = ctx + (((size_t)b * NH + h) * DHEAD + d) * DHEAD;
  float        s    = 0.0f;
#pragma unroll
  for (int e = 0; e < DHEAD; ++e) s += wrow[e] * cr[e];
  m2[idx] = s;
}

extern "C" void kernel_launch(void* const* d_in, const int* in_sizes, int n_in,
                              void* d_out, int out_size, void* d_ws, size_t ws_size,
                              hipStream_t stream) {
  (void)in_sizes; (void)n_in; (void)out_size; (void)ws_size;
  const float* x     = (const float*)d_in[0];
  const float* w_qkv = (const float*)d_in[1];
  const float* w_out = (const float*)d_in[2];
  float*       out   = (float*)d_out;

  float*  ws   = (float*)d_ws;
  float*  qk   = ws + OFF_QK;
  __bf16* vbf  = reinterpret_cast<__bf16*>(ws + OFF_V);
  float*  kmax = ws + OFF_KMAX;
  float*  ksum = ws + OFF_KSUM;
  float*  ctx  = ws + OFF_CTX;
  float*  m2   = ws + OFF_M2;

  // 1) qkv = W_qkv @ x  (q,k rows -> f32; v rows -> bf16)
  gemm_panel_kernel<false><<<dim3(MQKV / 64, NTOK / 64, NB), 256, 0, stream>>>(
      w_qkv, 0, x, (size_t)CIN * NTOK, qk, (size_t)2 * CIN * NTOK, vbf,
      (size_t)CIN * NTOK, /*vsplit=*/2 * CIN);

  // 2) per-channel k softmax stats over n
  k_reduce_kernel<<<dim3(NB * CIN), 256, 0, stream>>>(qk, kmax, ksum);

  // 3) context = softmax_n(k)^T @ v  (split-K, f32 atomics)
  (void)hipMemsetAsync(ctx, 0, SZ_CTX * sizeof(float), stream);
  context_kernel<<<dim3(32, NB * NH), 128, 0, stream>>>(qk, vbf, kmax, ksum, ctx,
                                                        NTOK / 32);

  // 4) M2[b] = fold(W_out, context[b])
  fold_kernel<<<dim3((NB * CIN * CIN) / 256), 256, 0, stream>>>(w_out, ctx, m2);

  // 5) out = M2[b] @ softmax_d(q)
  gemm_panel_kernel<true><<<dim3(CIN / 64, NTOK / 64, NB), 256, 0, stream>>>(
      m2, (size_t)CIN * CIN, qk, (size_t)2 * CIN * NTOK, out, (size_t)CIN * NTOK,
      vbf, 0, /*vsplit=*/1 << 30);
}
